// SHRED_diffrax_7670811591015
// MI455X (gfx1250) — compile-verified, matrix operand order checked by reading, not run on hardware
//
#include <hip/hip_runtime.h>
#include <hip/hip_bf16.h>

// ---------------------------------------------------------------------------
// Problem dims (from reference)
// ---------------------------------------------------------------------------
#define B_   64
#define T_   512
#define I_   64
#define H_   512
#define G4   2048     // 4*H
#define D1_  1024
#define D2_  2048
#define O_   16384
#define NWG  32       // one WG per 16-wide hidden slice (512/16)
#define LSTM_THREADS 512

typedef __bf16 bfv16 __attribute__((ext_vector_type(16)));
typedef __bf16 bfv8  __attribute__((ext_vector_type(8)));
typedef float  fv8   __attribute__((ext_vector_type(8)));

// ---------------------------------------------------------------------------
// Workspace layout (bytes); all sizes multiples of 256 so offsets stay aligned
// ---------------------------------------------------------------------------
static constexpr size_t OFF_XT   = 0;
static constexpr size_t SZ_XT    = (size_t)T_ * B_ * I_ * 2;     // x transposed, bf16
static constexpr size_t OFF_WIH0 = OFF_XT + SZ_XT;
static constexpr size_t SZ_WIH0  = (size_t)G4 * I_ * 2;
static constexpr size_t OFF_WHH0 = OFF_WIH0 + SZ_WIH0;
static constexpr size_t SZ_WHH   = (size_t)G4 * H_ * 2;
static constexpr size_t OFF_WIH1 = OFF_WHH0 + SZ_WHH;
static constexpr size_t OFF_WHH1 = OFF_WIH1 + SZ_WHH;
static constexpr size_t OFF_W1   = OFF_WHH1 + SZ_WHH;
static constexpr size_t SZ_W1    = (size_t)D1_ * H_ * 2;
static constexpr size_t OFF_W2   = OFF_W1 + SZ_W1;
static constexpr size_t SZ_W2    = (size_t)D2_ * D1_ * 2;
static constexpr size_t OFF_W3   = OFF_W2 + SZ_W2;
static constexpr size_t SZ_W3    = (size_t)O_ * D2_ * 2;
static constexpr size_t OFF_B0   = OFF_W3 + SZ_W3;               // combined biases f32
static constexpr size_t OFF_B1   = OFF_B0 + (size_t)G4 * 4;
static constexpr size_t OFF_H0   = OFF_B1 + (size_t)G4 * 4;      // double-buffered bf16
static constexpr size_t SZ_H     = (size_t)2 * B_ * H_ * 2;
static constexpr size_t OFF_H1   = OFF_H0 + SZ_H;
static constexpr size_t OFF_Y1   = OFF_H1 + SZ_H;                // MLP act bf16
static constexpr size_t OFF_Y2   = OFF_Y1 + (size_t)B_ * D1_ * 2;
static constexpr size_t OFF_SYNC = OFF_Y2 + (size_t)B_ * D2_ * 2;

// ---------------------------------------------------------------------------
// LSTM LDS layout (dynamic shared memory, 224.5 KB of the WGP's 320 KB)
// ---------------------------------------------------------------------------
static constexpr int    TILE_ELEMS = 512;            // 32 lanes * 16 bf16
static constexpr int    KT_IH0 = I_ / 32;            // 2
static constexpr int    KT_HH  = H_ / 32;            // 16
static constexpr size_t SEC_IH0 = 0;                                  // 8 tiles
static constexpr size_t SEC_HH0 = SEC_IH0 + (size_t)4 * KT_IH0 * TILE_ELEMS;
static constexpr size_t SEC_IH1 = SEC_HH0 + (size_t)4 * KT_HH * TILE_ELEMS;
static constexpr size_t SEC_HH1 = SEC_IH1 + (size_t)4 * KT_HH * TILE_ELEMS;
static constexpr size_t WLDS_ELEMS = SEC_HH1 + (size_t)4 * KT_HH * TILE_ELEMS; // 200*512
static constexpr size_t GBUF_OFF = WLDS_ELEMS * 2;               // 204800 B
static constexpr size_t CBUF_OFF = GBUF_OFF + (size_t)4 * 4 * 16 * 16 * 4; // +16 KB
static constexpr size_t BBUF_OFF = CBUF_OFF + (size_t)2 * B_ * 16 * 4;     // +8 KB
static constexpr size_t SMEM_TOTAL = BBUF_OFF + (size_t)2 * 4 * 16 * 4;    // +512 B

// ---------------------------------------------------------------------------
// Small helpers
// ---------------------------------------------------------------------------
__device__ __forceinline__ float sigf(float x) { return 1.0f / (1.0f + __expf(-x)); }
__device__ __forceinline__ float tanh_f(float x) {
  float e = __expf(-2.0f * fabsf(x));
  float t = (1.0f - e) / (1.0f + e);
  return x >= 0.0f ? t : -t;
}

__device__ __forceinline__ bfv16 join16(bfv8 lo, bfv8 hi) {
  return __builtin_shufflevector(lo, hi, 0,1,2,3,4,5,6,7,8,9,10,11,12,13,14,15);
}

// A-matrix (16x32 bf16 tile) load from row-major activations A[rows][K].
// CDNA5 layout: lanes 0-15 row M=lane, K={0..7,16..23}; lanes 16-31 K={8..15,24..31}.
__device__ __forceinline__ bfv16 load_a(const __bf16* __restrict__ A, int K,
                                        int row, int k0, int khalf) {
  const __bf16* p = A + (size_t)row * K + k0 + khalf * 8;
  bfv8 lo = *(const bfv8*)(p);
  bfv8 hi = *(const bfv8*)(p + 16);
  return join16(lo, hi);
}

// B-matrix tile: each lane's 16 values are contiguous (32 bytes).
__device__ __forceinline__ bfv16 load_b(const __bf16* p) {
  bfv8 lo = *(const bfv8*)(p);
  bfv8 hi = *(const bfv8*)(p + 8);
  return join16(lo, hi);
}

__device__ __forceinline__ fv8 wmma_bf16(bfv16 a, bfv16 b, fv8 c) {
  return __builtin_amdgcn_wmma_f32_16x16x32_bf16(false, a, false, b,
                                                 (short)0, c, false, false);
}

// Device-wide sense-reversing barrier (counters zeroed each launch).
__device__ __forceinline__ void grid_sync(unsigned* cnt, unsigned* epoch, unsigned nwg) {
  __threadfence();
  __syncthreads();
  if (threadIdx.x == 0) {
    unsigned e = atomicAdd(epoch, 0u);
    unsigned prev = atomicAdd(cnt, 1u);
    if (prev == nwg - 1u) {
      atomicExch(cnt, 0u);
      __threadfence();
      atomicAdd(epoch, 1u);
    } else {
      while (atomicAdd(epoch, 0u) == e) { __builtin_amdgcn_s_sleep(2); }
    }
  }
  __syncthreads();
  __threadfence();
}

// ---------------------------------------------------------------------------
// Prep kernels
// ---------------------------------------------------------------------------
__global__ void zero32(unsigned* p, int nwords) {
  for (int i = blockIdx.x * blockDim.x + threadIdx.x; i < nwords;
       i += gridDim.x * blockDim.x)
    p[i] = 0u;
}

// x [B,T,I] f32 -> xT [T,B,I] bf16
__global__ void convert_x(const float* __restrict__ x, __bf16* __restrict__ xT) {
  const int total = T_ * B_ * I_;
  for (int idx = blockIdx.x * blockDim.x + threadIdx.x; idx < total;
       idx += gridDim.x * blockDim.x) {
    int i = idx & (I_ - 1);
    int b = (idx >> 6) & (B_ - 1);
    int t = idx >> 12;
    xT[idx] = (__bf16)x[(size_t)b * T_ * I_ + (size_t)t * I_ + i];
  }
}

// Pack W[N][K] f32 (used as out = act @ W^T) into the per-lane WMMA B layout:
// tile (nt, kt) -> 32 lanes x 16 bf16 contiguous per lane, kt-minor ordering.
__global__ void pack_w(const float* __restrict__ W, __bf16* __restrict__ out,
                       int N, int K) {
  const int KT = K >> 5, NT = N >> 4;
  const int total = NT * KT * 32;
  for (int idx = blockIdx.x * blockDim.x + threadIdx.x; idx < total;
       idx += gridDim.x * blockDim.x) {
    int lane = idx & 31;
    int tile = idx >> 5;
    int kt = tile % KT;
    int nt = tile / KT;
    int col = nt * 16 + (lane & 15);
    int khalf = lane >> 4;
    __bf16* dst = out + (size_t)idx * 16;
    const float* src = W + (size_t)col * K + kt * 32;
#pragma unroll
    for (int j = 0; j < 8; ++j) dst[j]     = (__bf16)src[khalf * 8 + j];
#pragma unroll
    for (int j = 0; j < 8; ++j) dst[8 + j] = (__bf16)src[16 + khalf * 8 + j];
  }
}

__global__ void vec_add(const float* __restrict__ a, const float* __restrict__ b,
                        float* __restrict__ o, int n) {
  for (int i = blockIdx.x * blockDim.x + threadIdx.x; i < n;
       i += gridDim.x * blockDim.x)
    o[i] = a[i] + b[i];
}

// ---------------------------------------------------------------------------
// LSTM persistent kernel. Each WG owns hidden slice [wg*16, wg*16+16) of both
// layers. Its weight tiles (200 KB) are staged into LDS once; gates go through
// a 16 KB LDS buffer; cell state c lives entirely in LDS. Only h crosses WGs
// (double-buffered in global, published by a device-wide barrier, 2 per step).
// ---------------------------------------------------------------------------
__device__ __forceinline__ void copy_sec(__bf16* dstLds,
                                         const __bf16* __restrict__ src,
                                         int wg, int KT) {
  const int vecsPerTile = TILE_ELEMS / 8;   // 64 x uint4 per tile
  const int total = 4 * KT * vecsPerTile;
  for (int w = threadIdx.x; w < total; w += LSTM_THREADS) {
    int tile = w >> 6;
    int off  = w & 63;
    int q  = tile / KT;
    int kt = tile - q * KT;
    const uint4* s = (const uint4*)(src +
        (size_t)((q * (H_ / 16) + wg) * KT + kt) * TILE_ELEMS);
    uint4* d = (uint4*)(dstLds + (size_t)tile * TILE_ELEMS);
    d[off] = s[off];
  }
}

__device__ __forceinline__ void layer_step(
    const __bf16* __restrict__ A1, int K1, const __bf16* wsec1,
    const __bf16* __restrict__ A2, int K2, const __bf16* wsec2,
    const float* bbufL, float* cbufL, __bf16* __restrict__ h_out,
    float* gbuf, int wg) {
  const int tid   = threadIdx.x;
  const int wave  = tid >> 5;        // 0..15
  const int lane  = tid & 31;
  const int m     = wave & 3;        // batch tile (M)
  const int q     = wave >> 2;       // gate quadrant: 0=i,1=f,2=g,3=o
  const int ncol  = lane & 15;
  const int khalf = lane >> 4;
  const int row   = m * 16 + ncol;

  fv8 acc = {0.f, 0.f, 0.f, 0.f, 0.f, 0.f, 0.f, 0.f};
  {
    const int KT = K1 >> 5;
    const __bf16* wb = wsec1 + ((size_t)q * KT * 32 + lane) * 16;
    for (int kt = 0; kt < KT; ++kt) {
      bfv16 a = load_a(A1, K1, row, kt * 32, khalf);
      bfv16 b = load_b(wb + (size_t)kt * TILE_ELEMS);   // LDS
      acc = wmma_bf16(a, b, acc);
    }
  }
  {
    const int KT = K2 >> 5;
    const __bf16* wb = wsec2 + ((size_t)q * KT * 32 + lane) * 16;
    for (int kt = 0; kt < KT; ++kt) {
      bfv16 a = load_a(A2, K2, row, kt * 32, khalf);
      bfv16 b = load_b(wb + (size_t)kt * TILE_ELEMS);   // LDS
      acc = wmma_bf16(a, b, acc);
    }
  }

  // D layout: lanes 0-15 -> N=lane, rows v; lanes 16-31 -> rows v+8.
  float* g = gbuf + ((size_t)(q * 4 + m) * 16) * 16;
#pragma unroll
  for (int v = 0; v < 8; ++v) g[(v + 8 * khalf) * 16 + ncol] = acc[v];
  __syncthreads();

  // Fused LSTM cell update on this WG's 16-wide hidden slice (c in LDS).
  for (int cell = tid; cell < B_ * 16; cell += LSTM_THREADS) {
    int b = cell >> 4, j = cell & 15;
    int mt = b >> 4, r = b & 15;
    int col = wg * 16 + j;
    float iv = sigf  (gbuf[((0 * 4 + mt) * 16 + r) * 16 + j] + bbufL[0 * 16 + j]);
    float fv = sigf  (gbuf[((1 * 4 + mt) * 16 + r) * 16 + j] + bbufL[1 * 16 + j]);
    float gv = tanh_f(gbuf[((2 * 4 + mt) * 16 + r) * 16 + j] + bbufL[2 * 16 + j]);
    float ov = sigf  (gbuf[((3 * 4 + mt) * 16 + r) * 16 + j] + bbufL[3 * 16 + j]);
    float cp = cbufL[cell];
    float cn = fv * cp + iv * gv;
    cbufL[cell] = cn;
    h_out[(size_t)b * H_ + col] = (__bf16)(ov * tanh_f(cn));
  }
  __syncthreads();  // gbuf reusable afterwards
}

__global__ __launch_bounds__(LSTM_THREADS, 1) void lstm_kernel(
    const __bf16* __restrict__ xT,
    const __bf16* __restrict__ Wih0p, const __bf16* __restrict__ Whh0p,
    const __bf16* __restrict__ Wih1p, const __bf16* __restrict__ Whh1p,
    const float* __restrict__ bias0, const float* __restrict__ bias1,
    __bf16* __restrict__ h0, __bf16* __restrict__ h1, unsigned* sync) {
  extern __shared__ char smem_raw[];
  __bf16* wlds = (__bf16*)smem_raw;
  float*  gbuf = (float*)(smem_raw + GBUF_OFF);
  float*  cbuf = (float*)(smem_raw + CBUF_OFF);   // [2][B_*16]
  float*  bbuf = (float*)(smem_raw + BBUF_OFF);   // [2][4][16]
  const int wg = blockIdx.x;
  unsigned* cnt = sync;
  unsigned* epoch = sync + 16;

  // Stage this WG's weight tiles into LDS (once) + init c and bias slices.
  copy_sec(wlds + SEC_IH0, Wih0p, wg, KT_IH0);
  copy_sec(wlds + SEC_HH0, Whh0p, wg, KT_HH);
  copy_sec(wlds + SEC_IH1, Wih1p, wg, KT_HH);
  copy_sec(wlds + SEC_HH1, Whh1p, wg, KT_HH);
  for (int i = threadIdx.x; i < 2 * B_ * 16; i += LSTM_THREADS) cbuf[i] = 0.0f;
  for (int i = threadIdx.x; i < 2 * 4 * 16; i += LSTM_THREADS) {
    int l = i >> 6, idx = i & 63;
    const float* bl = l ? bias1 : bias0;
    bbuf[i] = bl[(idx >> 4) * H_ + wg * 16 + (idx & 15)];
  }
  __syncthreads();

  for (int t = 0; t < T_; ++t) {
    const int rp = t & 1, wp = rp ^ 1;
    asm volatile("" ::: "memory");  // keep LDS/global loads inside the t-loop
    // layer 0: gates = x_t @ Wih0^T + h0 @ Whh0^T
    layer_step(xT + (size_t)t * B_ * I_, I_, wlds + SEC_IH0,
               h0 + (size_t)rp * B_ * H_, H_, wlds + SEC_HH0,
               bbuf, cbuf, h0 + (size_t)wp * B_ * H_, gbuf, wg);
    grid_sync(cnt, epoch, gridDim.x);
    asm volatile("" ::: "memory");
    // layer 1: gates = h0(t) @ Wih1^T + h1 @ Whh1^T
    layer_step(h0 + (size_t)wp * B_ * H_, H_, wlds + SEC_IH1,
               h1 + (size_t)rp * B_ * H_, H_, wlds + SEC_HH1,
               bbuf + 64, cbuf + B_ * 16, h1 + (size_t)wp * B_ * H_, gbuf, wg);
    grid_sync(cnt, epoch, gridDim.x);
  }
}

// ---------------------------------------------------------------------------
// MLP GEMM: out[64][N] = act(A[64][K] @ W^T + bias); one WMMA tile per wave.
// Weights stream from global (single pass; W3 is HBM-bandwidth bound).
// ---------------------------------------------------------------------------
__global__ __launch_bounds__(256, 2) void gemm_kernel(
    const __bf16* __restrict__ A, const __bf16* __restrict__ Wp,
    const float* __restrict__ bias, int N, int K,
    float* __restrict__ outF, __bf16* __restrict__ outB, int relu) {
  const int wave  = threadIdx.x >> 5;
  const int lane  = threadIdx.x & 31;
  const int ncol  = lane & 15;
  const int khalf = lane >> 4;
  const int KT = K >> 5, NT = N >> 4;
  const int tiles = 4 * NT;  // M = 64 -> 4 M-tiles
  const int wstep = (blockDim.x >> 5) * gridDim.x;

  for (int tile = blockIdx.x * (blockDim.x >> 5) + wave; tile < tiles;
       tile += wstep) {
    int m = tile & 3, nt = tile >> 2;
    int row = m * 16 + ncol;
    fv8 acc = {0.f, 0.f, 0.f, 0.f, 0.f, 0.f, 0.f, 0.f};
    const __bf16* wbase = Wp + ((size_t)nt * KT * 32 + lane) * 16;
    for (int kt = 0; kt < KT; ++kt) {
      bfv16 a = load_a(A, K, row, kt * 32, khalf);
      const __bf16* p = wbase + (size_t)kt * TILE_ELEMS;
      __builtin_prefetch(p + 2 * TILE_ELEMS, 0, 1);   // stream next K-tiles
      bfv16 b = load_b(p);
      acc = wmma_bf16(a, b, acc);
    }
    int colg = nt * 16 + ncol;
    float bb = bias[colg];
#pragma unroll
    for (int v = 0; v < 8; ++v) {
      int r = m * 16 + v + 8 * khalf;
      float val = acc[v] + bb;
      if (relu) val = fmaxf(val, 0.0f);
      if (outF) outF[(size_t)r * N + colg] = val;
      else      outB[(size_t)r * N + colg] = (__bf16)val;
    }
  }
}

// ---------------------------------------------------------------------------
// Host launch
// ---------------------------------------------------------------------------
extern "C" void kernel_launch(void* const* d_in, const int* in_sizes, int n_in,
                              void* d_out, int out_size, void* d_ws, size_t ws_size,
                              hipStream_t stream) {
  (void)in_sizes; (void)n_in; (void)out_size; (void)ws_size;

  const float* x    = (const float*)d_in[0];
  const float* Wih0 = (const float*)d_in[1];
  const float* Whh0 = (const float*)d_in[2];
  const float* bih0 = (const float*)d_in[3];
  const float* bhh0 = (const float*)d_in[4];
  const float* Wih1 = (const float*)d_in[5];
  const float* Whh1 = (const float*)d_in[6];
  const float* bih1 = (const float*)d_in[7];
  const float* bhh1 = (const float*)d_in[8];
  const float* W1   = (const float*)d_in[9];
  const float* b1   = (const float*)d_in[10];
  const float* W2   = (const float*)d_in[11];
  const float* b2   = (const float*)d_in[12];
  const float* W3   = (const float*)d_in[13];
  const float* b3   = (const float*)d_in[14];

  char* ws = (char*)d_ws;
  __bf16* xT    = (__bf16*)(ws + OFF_XT);
  __bf16* Wih0p = (__bf16*)(ws + OFF_WIH0);
  __bf16* Whh0p = (__bf16*)(ws + OFF_WHH0);
  __bf16* Wih1p = (__bf16*)(ws + OFF_WIH1);
  __bf16* Whh1p = (__bf16*)(ws + OFF_WHH1);
  __bf16* W1p   = (__bf16*)(ws + OFF_W1);
  __bf16* W2p   = (__bf16*)(ws + OFF_W2);
  __bf16* W3p   = (__bf16*)(ws + OFF_W3);
  float*  bias0 = (float*)(ws + OFF_B0);
  float*  bias1 = (float*)(ws + OFF_B1);
  __bf16* h0    = (__bf16*)(ws + OFF_H0);
  __bf16* h1    = (__bf16*)(ws + OFF_H1);
  __bf16* y1    = (__bf16*)(ws + OFF_Y1);
  __bf16* y2    = (__bf16*)(ws + OFF_Y2);
  unsigned* sync = (unsigned*)(ws + OFF_SYNC);

  // Allow 224.5 KB dynamic LDS for the LSTM kernel (CDNA5 WGP has 320 KB).
  hipFuncSetAttribute((const void*)lstm_kernel,
                      hipFuncAttributeMaxDynamicSharedMemorySize,
                      (int)SMEM_TOTAL);

  // Zero recurrent h buffers and barrier counters.
  zero32<<<256, 256, 0, stream>>>((unsigned*)(ws + OFF_H0), (int)(2 * SZ_H / 4));
  zero32<<<1, 64, 0, stream>>>(sync, 64);

  // Precision/layout conversion: f32 -> bf16, weights swizzled to WMMA-B tiles.
  convert_x<<<2048, 256, 0, stream>>>(x, xT);
  pack_w<<<1024, 256, 0, stream>>>(Wih0, Wih0p, G4, I_);
  pack_w<<<1024, 256, 0, stream>>>(Whh0, Whh0p, G4, H_);
  pack_w<<<1024, 256, 0, stream>>>(Wih1, Wih1p, G4, H_);
  pack_w<<<1024, 256, 0, stream>>>(Whh1, Whh1p, G4, H_);
  pack_w<<<1024, 256, 0, stream>>>(W1, W1p, D1_, H_);
  pack_w<<<1024, 256, 0, stream>>>(W2, W2p, D2_, D1_);
  pack_w<<<2048, 256, 0, stream>>>(W3, W3p, O_, D2_);
  vec_add<<<8, 256, 0, stream>>>(bih0, bhh0, bias0, G4);
  vec_add<<<8, 256, 0, stream>>>(bih1, bhh1, bias1, G4);

  // Persistent 512-step double-layer LSTM (2 grid barriers per step),
  // weights LDS-resident.
  lstm_kernel<<<NWG, LSTM_THREADS, SMEM_TOTAL, stream>>>(
      xT, Wih0p, Whh0p, Wih1p, Whh1p, bias0, bias1, h0, h1, sync);

  // Decoder MLP on final h1 (in buffer 0 after t=511).
  gemm_kernel<<<32, 256, 0, stream>>>(h1, W1p, b1, D1_, H_, nullptr, y1, 1);
  gemm_kernel<<<64, 256, 0, stream>>>(y1, W2p, b2, D2_, D1_, nullptr, y2, 1);
  gemm_kernel<<<512, 256, 0, stream>>>(y2, W3p, b3, O_, D2_, (float*)d_out,
                                       nullptr, 0);
}
